// HeteroGATv2_2662879723775
// MI455X (gfx1250) — compile-verified, matrix operand order checked by reading, not run on hardware
//
#include <hip/hip_runtime.h>
#include <limits.h>

// ---------------------------------------------------------------------------
// Types for CDNA5 WMMA (wave32, gfx1250)
// ---------------------------------------------------------------------------
typedef __bf16 bf16_t;
typedef bf16_t v16bf __attribute__((ext_vector_type(16)));
typedef bf16_t v8bf  __attribute__((ext_vector_type(8)));
typedef bf16_t v4bf  __attribute__((ext_vector_type(4)));
typedef bf16_t v2bf  __attribute__((ext_vector_type(2)));
typedef float  v8f   __attribute__((ext_vector_type(8)));

union FragU { v16bf v; v8bf h[2]; };

// ---------------------------------------------------------------------------
// Tiled bf16 WMMA GEMM:  C[M,N] = A[M,K] * B[K,N] + bias[N]  (optional ReLU)
//
// Block = 256 threads (8 waves, wave32). Tile 128(M) x 64(N), K-step 32.
// Waves arranged 4(M) x 2(N); each wave owns 32x32 = 2x2 WMMA fragments
// -> 4 x v_wmma_f32_16x16x32_bf16 per K-step with 2x register reuse.
//
// LDS double-buffered; one barrier per K-step. Fragment-friendly layout
// (ISA 7.12.2: lane's 16 elements at k = k0..k0+7 and k0+16..k0+23,
//  k0=(lane>=16)*8):
//   As: row-major  [128][40]  (row stride 80B, 16B aligned -> 2x ds_load_b128)
//   Bs: TRANSPOSED [ 64][40]  (N-major, K-minor -> same 2x ds_load_b128 shape)
// ---------------------------------------------------------------------------
#define BM 128
#define BN 64
#define BK 32
#define TSTRIDE 40   // halfwords per LDS row: 80 bytes, 16B-aligned, bank-spread
#define ASZ (BM * TSTRIDE)
#define BSZ (BN * TSTRIDE)

__device__ __forceinline__ void stageA_fast(const float* __restrict__ A, bf16_t* as,
                                            int blockM, int K, int kk, int tid)
{
    #pragma unroll
    for (int q = 0; q < 4; ++q) {
        int li = tid + q * 256;
        int r = li >> 3, c4 = (li & 7) << 2;
        float4 v = *(const float4*)&A[(size_t)(blockM + r) * K + kk + c4];
        v4bf p;
        p[0] = (bf16_t)v.x; p[1] = (bf16_t)v.y;
        p[2] = (bf16_t)v.z; p[3] = (bf16_t)v.w;
        *(v4bf*)&as[r * TSTRIDE + c4] = p;           // ds_store_b64
    }
}

__device__ __forceinline__ void stageA_slow(const float* __restrict__ A, bf16_t* as,
                                            int blockM, int M, int K, int kk, int tid)
{
    #pragma unroll
    for (int q = 0; q < 16; ++q) {
        int li = tid + q * 256;
        int r = li >> 5, c = li & 31;
        int gm = blockM + r, gk = kk + c;
        float v = (gm < M && gk < K) ? A[(size_t)gm * K + gk] : 0.0f;
        as[r * TSTRIDE + c] = (bf16_t)v;
    }
}

__device__ __forceinline__ void stageB_fast(const float* __restrict__ B, bf16_t* bs,
                                            int blockN, int N, int kk, int tid)
{
    #pragma unroll
    for (int q = 0; q < 4; ++q) {
        int li = tid + q * 256;
        int n  = li & 63;
        int k2 = (li >> 6) << 1;
        int gn = blockN + n;
        float v0 = B[(size_t)(kk + k2)     * N + gn];
        float v1 = B[(size_t)(kk + k2 + 1) * N + gn];
        v2bf p; p[0] = (bf16_t)v0; p[1] = (bf16_t)v1;
        *(v2bf*)&bs[n * TSTRIDE + k2] = p;           // ds_store_b32
    }
}

__device__ __forceinline__ void stageB_slow(const float* __restrict__ B, bf16_t* bs,
                                            int blockN, int N, int K, int kk, int tid)
{
    #pragma unroll
    for (int q = 0; q < 8; ++q) {
        int li = tid + q * 256;
        int r = li >> 6, c = li & 63;   // r = k, c = n
        int gk = kk + r, gn = blockN + c;
        float v = (gk < K && gn < N) ? B[(size_t)gk * N + gn] : 0.0f;
        bs[c * TSTRIDE + r] = (bf16_t)v;
    }
}

__device__ __forceinline__ void frag_mma(const bf16_t* as, const bf16_t* bs,
                                         int wm, int wn, int l15, int k0,
                                         v8f& acc00, v8f& acc01, v8f& acc10, v8f& acc11)
{
    FragU a0, a1, b0, b1;
    const bf16_t* pa0 = &as[(wm * 32 + l15)      * TSTRIDE + k0];
    const bf16_t* pa1 = &as[(wm * 32 + 16 + l15) * TSTRIDE + k0];
    const bf16_t* pb0 = &bs[(wn * 32 + l15)      * TSTRIDE + k0];
    const bf16_t* pb1 = &bs[(wn * 32 + 16 + l15) * TSTRIDE + k0];
    a0.h[0] = *(const v8bf*)(pa0);      a0.h[1] = *(const v8bf*)(pa0 + 16);
    a1.h[0] = *(const v8bf*)(pa1);      a1.h[1] = *(const v8bf*)(pa1 + 16);
    b0.h[0] = *(const v8bf*)(pb0);      b0.h[1] = *(const v8bf*)(pb0 + 16);
    b1.h[0] = *(const v8bf*)(pb1);      b1.h[1] = *(const v8bf*)(pb1 + 16);
    acc00 = __builtin_amdgcn_wmma_f32_16x16x32_bf16(false, a0.v, false, b0.v, (short)0, acc00, false, false);
    acc01 = __builtin_amdgcn_wmma_f32_16x16x32_bf16(false, a0.v, false, b1.v, (short)0, acc01, false, false);
    acc10 = __builtin_amdgcn_wmma_f32_16x16x32_bf16(false, a1.v, false, b0.v, (short)0, acc10, false, false);
    acc11 = __builtin_amdgcn_wmma_f32_16x16x32_bf16(false, a1.v, false, b1.v, (short)0, acc11, false, false);
}

__global__ __launch_bounds__(256) void gemm_bias_wmma(
    const float* __restrict__ A, const float* __restrict__ B,
    const float* __restrict__ bias, float* __restrict__ C,
    int M, int N, int K, int relu)
{
    __shared__ bf16_t As[2][ASZ];
    __shared__ bf16_t Bs[2][BSZ];

    const int tid  = threadIdx.x;
    const int lane = tid & 31;
    const int wave = tid >> 5;
    const int wm   = wave & 3;       // 0..3 -> 32-row slice
    const int wn   = wave >> 2;      // 0..1 -> 32-col slice
    const int blockM = blockIdx.y * BM;
    const int blockN = blockIdx.x * BN;
    const int l15 = lane & 15;
    const int k0  = (lane >> 4) << 3;          // 0 or 8

    v8f acc00 = {}, acc01 = {}, acc10 = {}, acc11 = {};

    const bool interior = ((blockM + BM) <= M) && ((blockN + BN) <= N);
    const int  nTiles   = (K + BK - 1) / BK;

    // ---- prologue: stage tile 0 into buffer 0 ----
    if (interior && BK <= K) {
        stageA_fast(A, As[0], blockM, K, 0, tid);
        stageB_fast(B, Bs[0], blockN, N, 0, tid);
    } else {
        stageA_slow(A, As[0], blockM, M, K, 0, tid);
        stageB_slow(B, Bs[0], blockN, N, K, 0, tid);
    }
    __syncthreads();

    // ---- main software-pipelined loop: one barrier per K-step ----
    if (interior) {
        for (int t = 0; t < nTiles; ++t) {
            const int cur = t & 1;
            const int kkn = (t + 1) * BK;
            if (t + 1 < nTiles) {
                if (kkn + BK <= K) {
                    stageA_fast(A, As[1 - cur], blockM, K, kkn, tid);
                    stageB_fast(B, Bs[1 - cur], blockN, N, kkn, tid);
                } else {           // K-tail tile (only for K % 32 != 0)
                    stageA_slow(A, As[1 - cur], blockM, M, K, kkn, tid);
                    stageB_slow(B, Bs[1 - cur], blockN, N, K, kkn, tid);
                }
            }
            frag_mma(As[cur], Bs[cur], wm, wn, l15, k0, acc00, acc01, acc10, acc11);
            __syncthreads();
        }
    } else {
        for (int t = 0; t < nTiles; ++t) {
            const int cur = t & 1;
            const int kkn = (t + 1) * BK;
            if (t + 1 < nTiles) {
                stageA_slow(A, As[1 - cur], blockM, M, K, kkn, tid);
                stageB_slow(B, Bs[1 - cur], blockN, N, K, kkn, tid);
            }
            frag_mma(As[cur], Bs[cur], wm, wn, l15, k0, acc00, acc01, acc10, acc11);
            __syncthreads();
        }
    }

    // ---- store: D row = vgpr + (lane/16)*8 within 16x16 tile, col = lane%16
    const int rbase = blockM + wm * 32 + ((lane >> 4) << 3);
    const int cbase = blockN + wn * 32 + l15;
    #pragma unroll
    for (int r = 0; r < 8; ++r) {
        int gm0 = rbase + r;        // m-frag 0
        int gm1 = gm0 + 16;         // m-frag 1
        int c0 = cbase, c1 = cbase + 16;
        if (gm0 < M) {
            if (c0 < N) {
                float v = acc00[r] + bias[c0];
                if (relu) v = fmaxf(v, 0.0f);
                C[(size_t)gm0 * N + c0] = v;
            }
            if (c1 < N) {
                float v = acc01[r] + bias[c1];
                if (relu) v = fmaxf(v, 0.0f);
                C[(size_t)gm0 * N + c1] = v;
            }
        }
        if (gm1 < M) {
            if (c0 < N) {
                float v = acc10[r] + bias[c0];
                if (relu) v = fmaxf(v, 0.0f);
                C[(size_t)gm1 * N + c0] = v;
            }
            if (c1 < N) {
                float v = acc11[r] + bias[c1];
                if (relu) v = fmaxf(v, 0.0f);
                C[(size_t)gm1 * N + c1] = v;
            }
        }
    }
}

// ---------------------------------------------------------------------------
// Utility fills / elementwise kernels
// ---------------------------------------------------------------------------
__global__ void fill_f32(float* __restrict__ p, float v, size_t n) {
    size_t i = (size_t)blockIdx.x * blockDim.x + threadIdx.x;
    if (i < n) p[i] = v;
}
__global__ void fill_i32(int* __restrict__ p, int v, size_t n) {
    size_t i = (size_t)blockIdx.x * blockDim.x + threadIdx.x;
    if (i < n) p[i] = v;
}
__global__ void combine_half(const float* __restrict__ a, const float* __restrict__ b,
                             float* __restrict__ o, size_t n) {
    size_t i = (size_t)blockIdx.x * blockDim.x + threadIdx.x;
    if (i < n) o[i] = 0.5f * (a[i] + b[i]);
}
__global__ void copy_f32(const float* __restrict__ a, float* __restrict__ o, size_t n) {
    size_t i = (size_t)blockIdx.x * blockDim.x + threadIdx.x;
    if (i < n) o[i] = a[i];
}

// ---------------------------------------------------------------------------
// Edge phase: GATv2 segment softmax (H=4 heads, D=32 per head, HD=128)
// ---------------------------------------------------------------------------
__device__ __forceinline__ int fmap_ord(float f) {   // monotonic float->int map
    int i = __float_as_int(f);
    return (i >= 0) ? i : (i ^ 0x7FFFFFFF);
}
__device__ __forceinline__ float funmap_ord(int i) {
    return __int_as_float((i >= 0) ? i : (i ^ 0x7FFFFFFF));
}

// score_eh = dot_d( leaky_relu(fs[src]+fd[dst], 0.2), attn[h] );  + seg-max
__global__ void edge_score_max(const int* __restrict__ src, const int* __restrict__ dst,
                               const float* __restrict__ fs, const float* __restrict__ fd,
                               const float* __restrict__ attn,
                               float* __restrict__ esc, int* __restrict__ mI, int E)
{
    int idx = blockIdx.x * blockDim.x + threadIdx.x;
    if (idx >= E * 4) return;
    int e = idx >> 2, h = idx & 3;
    int s = src[e], d = dst[e];
    const float* pf = fs + (size_t)s * 128 + h * 32;
    const float* pd = fd + (size_t)d * 128 + h * 32;
    const float* pa = attn + h * 32;
    float acc = 0.0f;
    #pragma unroll
    for (int k = 0; k < 32; ++k) {
        float v = pf[k] + pd[k];
        v = (v > 0.0f) ? v : 0.2f * v;
        acc = fmaf(v, pa[k], acc);
    }
    esc[idx] = acc;
    atomicMax(&mI[(size_t)d * 4 + h], fmap_ord(acc));
}

__global__ void edge_exp_den(const int* __restrict__ dst, float* __restrict__ esc,
                             const int* __restrict__ mI, float* __restrict__ den, int E)
{
    int idx = blockIdx.x * blockDim.x + threadIdx.x;
    if (idx >= E * 4) return;
    int e = idx >> 2, h = idx & 3;
    int d = dst[e];
    float m  = funmap_ord(mI[(size_t)d * 4 + h]);
    float ex = __expf(esc[idx] - m);
    esc[idx] = ex;
    atomicAdd(&den[(size_t)d * 4 + h], ex);
}

// out[dst, r] += (ex/den) * fs[src, r], thread per (edge, feature)
__global__ void edge_scatter(const int* __restrict__ src, const int* __restrict__ dst,
                             const float* __restrict__ esc, const float* __restrict__ den,
                             const float* __restrict__ fs, float* __restrict__ out, int E)
{
    size_t idx = (size_t)blockIdx.x * blockDim.x + threadIdx.x;
    if (idx >= (size_t)E * 128) return;
    int e = (int)(idx >> 7), r = (int)(idx & 127), h = r >> 5;
    int s = src[e], d = dst[e];
    float a = esc[(size_t)e * 4 + h] / den[(size_t)d * 4 + h];
    atomicAdd(&out[(size_t)d * 128 + r], a * fs[(size_t)s * 128 + r]);
}

// classifier tail: logits = hcls(ND x 64) @ Wc2(64x2) + bc2
__global__ void classifier_tail(const float* __restrict__ hcls, const float* __restrict__ Wc2,
                                const float* __restrict__ bc2, float* __restrict__ out, int M)
{
    int i = blockIdx.x * blockDim.x + threadIdx.x;
    if (i >= M) return;
    const float* h = hcls + (size_t)i * 64;
    float a0 = bc2[0], a1 = bc2[1];
    #pragma unroll
    for (int k = 0; k < 64; ++k) {
        float v = h[k];
        a0 = fmaf(v, Wc2[k * 2 + 0], a0);
        a1 = fmaf(v, Wc2[k * 2 + 1], a1);
    }
    out[i * 2 + 0] = a0;
    out[i * 2 + 1] = a1;
}

// ---------------------------------------------------------------------------
// Host orchestration
// ---------------------------------------------------------------------------
extern "C" void kernel_launch(void* const* d_in, const int* in_sizes, int n_in,
                              void* d_out, int out_size, void* d_ws, size_t ws_size,
                              hipStream_t stream)
{
    (void)n_in; (void)out_size; (void)ws_size;
    const int NW = 100000, ND = 50000;

    const float* x_word = (const float*)d_in[0];
    const float* x_doc  = (const float*)d_in[1];
    const int* src_w2d = (const int*)d_in[2];
    const int* dst_w2d = (const int*)d_in[3];
    const int* src_d2w = (const int*)d_in[4];
    const int* dst_d2w = (const int*)d_in[5];
    const int* src_cit = (const int*)d_in[6];
    const int* dst_cit = (const int*)d_in[7];
    const float* Wp_word = (const float*)d_in[8];
    const float* bp_word = (const float*)d_in[9];
    const float* Wp_doc  = (const float*)d_in[10];
    const float* bp_doc  = (const float*)d_in[11];
    const float* WsL[2] = {(const float*)d_in[12], (const float*)d_in[17]};
    const float* bsL[2] = {(const float*)d_in[13], (const float*)d_in[18]};
    const float* WdL[2] = {(const float*)d_in[14], (const float*)d_in[19]};
    const float* bdL[2] = {(const float*)d_in[15], (const float*)d_in[20]};
    const float* atL[2] = {(const float*)d_in[16], (const float*)d_in[21]};
    const float* Wc1 = (const float*)d_in[22];
    const float* bc1 = (const float*)d_in[23];
    const float* Wc2 = (const float*)d_in[24];
    const float* bc2 = (const float*)d_in[25];

    const int E_w2d = in_sizes[2];
    const int E_d2w = in_sizes[4];
    const int E_cit = in_sizes[6];

    // ---- workspace layout (floats) ----
    float* ws = (float*)d_ws;
    size_t off = 0;
    auto alloc = [&](size_t n) { float* p = ws + off; off += (n + 63) & ~(size_t)63; return p; };
    float* hw   = alloc((size_t)NW * 256);
    float* hd   = alloc((size_t)ND * 256);
    float* fsW  = alloc((size_t)NW * 128);
    float* fdW  = alloc((size_t)ND * 128);
    float* fsD  = alloc((size_t)ND * 128);
    float* fdD  = alloc((size_t)NW * 128);
    float* fsC  = alloc((size_t)ND * 128);
    float* fdC  = alloc((size_t)ND * 128);
    float* oW   = alloc((size_t)ND * 128);
    float* oD   = alloc((size_t)NW * 128);
    float* oC   = alloc((size_t)ND * 128);
    float* esc  = alloc((size_t)800000 * 4);
    float* den  = alloc((size_t)NW * 4);
    int*   mI   = (int*)alloc((size_t)NW * 4);
    float* hcls = alloc((size_t)ND * 64);

    auto gemm = [&](const float* A, const float* B, const float* bias, float* C,
                    int M, int N, int K, int relu) {
        dim3 grid((N + BN - 1) / BN, (M + BM - 1) / BM);
        gemm_bias_wmma<<<grid, 256, 0, stream>>>(A, B, bias, C, M, N, K, relu);
    };
    auto gat_edges = [&](const int* src, const int* dst, int E,
                         const float* fs, const float* fd, const float* attn,
                         float* out, int Ndst) {
        size_t nH = (size_t)Ndst * 4, nO = (size_t)Ndst * 128;
        fill_i32<<<(unsigned)((nH + 255) / 256), 256, 0, stream>>>(mI, INT_MIN, nH);
        fill_f32<<<(unsigned)((nH + 255) / 256), 256, 0, stream>>>(den, 0.0f, nH);
        fill_f32<<<(unsigned)((nO + 255) / 256), 256, 0, stream>>>(out, 0.0f, nO);
        int nEH = E * 4;
        edge_score_max<<<(nEH + 255) / 256, 256, 0, stream>>>(src, dst, fs, fd, attn, esc, mI, E);
        edge_exp_den<<<(nEH + 255) / 256, 256, 0, stream>>>(dst, esc, mI, den, E);
        size_t nS = (size_t)E * 128;
        edge_scatter<<<(unsigned)((nS + 255) / 256), 256, 0, stream>>>(src, dst, esc, den, fs, out, E);
    };

    // ---- feature projection ----
    gemm(x_word, Wp_word, bp_word, hw, NW, 256, 300, 0);
    gemm(x_doc,  Wp_doc,  bp_doc,  hd, ND, 256, 256, 0);

    // ---- two GATv2 layers ----
    for (int l = 0; l < 2; ++l) {
        const int Kin = (l == 0) ? 256 : 128;
        const float *Ws = WsL[l], *bs = bsL[l], *Wd = WdL[l], *bd = bdL[l], *at = atL[l];
        const size_t wstride = (size_t)Kin * 128;
        // fs/fd for etypes: 0=w2d(words->docs) 1=d2w(docs->words) 2=cites(docs->docs)
        gemm(hw, Ws + 0 * wstride, bs + 0 * 128, fsW, NW, 128, Kin, 0);
        gemm(hd, Wd + 0 * wstride, bd + 0 * 128, fdW, ND, 128, Kin, 0);
        gemm(hd, Ws + 1 * wstride, bs + 1 * 128, fsD, ND, 128, Kin, 0);
        gemm(hw, Wd + 1 * wstride, bd + 1 * 128, fdD, NW, 128, Kin, 0);
        gemm(hd, Ws + 2 * wstride, bs + 2 * 128, fsC, ND, 128, Kin, 0);
        gemm(hd, Wd + 2 * wstride, bd + 2 * 128, fdC, ND, 128, Kin, 0);

        gat_edges(src_w2d, dst_w2d, E_w2d, fsW, fdW, at + 0 * 128, oW, ND);
        gat_edges(src_d2w, dst_d2w, E_d2w, fsD, fdD, at + 1 * 128, oD, NW);
        gat_edges(src_cit, dst_cit, E_cit, fsC, fdC, at + 2 * 128, oC, ND);

        size_t nHd = (size_t)ND * 128, nHw = (size_t)NW * 128;
        combine_half<<<(unsigned)((nHd + 255) / 256), 256, 0, stream>>>(oW, oC, hd, nHd);
        copy_f32<<<(unsigned)((nHw + 255) / 256), 256, 0, stream>>>(oD, hw, nHw);
    }

    // ---- classifier ----
    gemm(hd, Wc1, bc1, hcls, ND, 64, 128, 1);
    classifier_tail<<<(ND + 255) / 256, 256, 0, stream>>>(hcls, Wc2, bc2, (float*)d_out, ND);
}
